// MultiHeadEmotionalAttention_39633958207833
// MI455X (gfx1250) — matrix operand order repaired and checked by reading, not compiled, not run on hardware
//
#include <hip/hip_runtime.h>
#include <hip/hip_bf16.h>

// ---------------------------------------------------------------------------
// MultiHeadEmotionalAttention for MI455X (gfx1250, wave32, WMMA).
// All matmuls use v_wmma_f32_16x16x32_bf16 (bf16 in, f32 accumulate).
// B=4, S=2048, E=512, H=8, D=64.
// ---------------------------------------------------------------------------

#define B_ 4
#define S_ 2048
#define E_ 512
#define H_ 8
#define D_ 64

typedef __attribute__((ext_vector_type(16))) __bf16 v16bf;
typedef __attribute__((ext_vector_type(8)))  __bf16 v8bf;
typedef __attribute__((ext_vector_type(8)))  float  v8f;

// Native converts: lower to v_cvt_pk_bf16_f32 on gfx1250 (RNE).
static __device__ __forceinline__ __bf16 f2bf(float f) { return (__bf16)f; }
static __device__ __forceinline__ float  bf2f(__bf16 b) { return (float)b; }

static __device__ __forceinline__ v8bf ld8(const __bf16* p) {
  return *reinterpret_cast<const v8bf*>(p); // 16B-aligned by construction
}
static __device__ __forceinline__ void st8(__bf16* p, v8bf v) {
  *reinterpret_cast<v8bf*>(p) = v;
}
static __device__ __forceinline__ v16bf cat16(v8bf lo, v8bf hi) {
  return __builtin_shufflevector(lo, hi, 0,1,2,3,4,5,6,7,8,9,10,11,12,13,14,15);
}
static __device__ __forceinline__ v8f wmma_bf(v16bf a, v16bf b, v8f c) {
  // (neg_a, A, neg_b, B, c_mod, C, reuse_a, reuse_b)
  return __builtin_amdgcn_wmma_f32_16x16x32_bf16(false, a, false, b, (short)0, c,
                                                 false, false);
}
// A-fragment (16x32, 16-bit): lane half h: elems 0..7 -> K=8h+e, 8..15 -> K=16+8h+e
static __device__ __forceinline__ v16bf afrag_bf(const __bf16* p, int h) {
  return cat16(ld8(p + 8 * h), ld8(p + 16 + 8 * h));
}
// B-fragment (32x16, 16-bit): lane half h: elems 0..7 -> K=8h+e, 8..15 -> K=16+8h+e
static __device__ __forceinline__ v16bf bfrag_bf(const __bf16* p, int h) {
  return cat16(ld8(p + 8 * h), ld8(p + 16 + 8 * h));
}
static __device__ __forceinline__ v8f vzero() {
  v8f z = {0.f, 0.f, 0.f, 0.f, 0.f, 0.f, 0.f, 0.f};
  return z;
}

// ---------------------------------------------------------------------------
// fp32 -> bf16 conversion, 4 elements/thread (b128 load, b64 store)
// ---------------------------------------------------------------------------
__global__ void cvt_bf16_kernel(const float* __restrict__ src,
                                __bf16* __restrict__ dst, int n4) {
  int i = blockIdx.x * blockDim.x + threadIdx.x;
  if (i < n4) {
    const float4 v = reinterpret_cast<const float4*>(src)[i];
    __bf16* o = dst + 4 * (size_t)i;
    o[0] = f2bf(v.x); o[1] = f2bf(v.y); o[2] = f2bf(v.z); o[3] = f2bf(v.w);
  }
}

// ---------------------------------------------------------------------------
// Y = X @ W^T + b : X bf16 [M=B*S, E], W bf16 [E,E] (rows dotted with rows).
// 2x2 register tile (32x32 output) per wave: 4 WMMAs per 4 fragment loads,
// 16 k-steps of K=32.  TRANSPOSE_V==0: dst=[B,H,S,D]; ==1: dst=[B,H,D,S].
// ---------------------------------------------------------------------------
template <int TRANSPOSE_V>
__global__ void proj_qkv_kernel(const __bf16* __restrict__ X,
                                const __bf16* __restrict__ Wb,
                                const float* __restrict__ bias,
                                __bf16* __restrict__ dst) {
  const int wave = threadIdx.x >> 5;
  const int lane = threadIdx.x & 31;
  const int h = lane >> 4, l15 = lane & 15;
  const int gid = blockIdx.x * 4 + wave;   // (M/32)*(E/32) = 4096 tiles
  const int tn = gid & 15;                 // E/32 = 16 column tiles
  const int tm = gid >> 4;                 // (B*S)/32 = 256 row tiles

  const __bf16* arow0 = X  + (size_t)(tm * 32 + l15) * E_;
  const __bf16* arow1 = arow0 + (size_t)16 * E_;
  const __bf16* brow0 = Wb + (size_t)(tn * 32 + l15) * E_;
  const __bf16* brow1 = brow0 + (size_t)16 * E_;

  v8f acc00 = vzero(), acc01 = vzero(), acc10 = vzero(), acc11 = vzero();
#pragma unroll 2
  for (int kc = 0; kc < E_ / 32; ++kc) {
    const int kb = kc * 32;
    v16bf a0 = afrag_bf(arow0 + kb, h);
    v16bf a1 = afrag_bf(arow1 + kb, h);
    v16bf b0 = bfrag_bf(brow0 + kb, h);
    v16bf b1 = bfrag_bf(brow1 + kb, h);
    acc00 = wmma_bf(a0, b0, acc00);
    acc01 = wmma_bf(a0, b1, acc01);
    acc10 = wmma_bf(a1, b0, acc10);
    acc11 = wmma_bf(a1, b1, acc11);
  }
#pragma unroll
  for (int j = 0; j < 2; ++j) {
    const int gcol = tn * 32 + j * 16 + l15;
    const float bv = bias[gcol];
    const int hh = gcol >> 6, d = gcol & 63;
#pragma unroll
    for (int i = 0; i < 2; ++i) {
      const v8f& acc = (i == 0) ? (j == 0 ? acc00 : acc01)
                                : (j == 0 ? acc10 : acc11);
#pragma unroll
      for (int r = 0; r < 8; ++r) {
        const int grow = tm * 32 + i * 16 + r + 8 * h;  // C: M = r + 8*half
        const int bt = grow >> 11, s = grow & 2047;
        const float y = acc[r] + bv;
        const size_t hb = (size_t)(bt * H_ + hh);
        if (TRANSPOSE_V) dst[(hb * D_ + d) * S_ + s] = f2bf(y);
        else             dst[(hb * S_ + s) * D_ + d] = f2bf(y);
      }
    }
  }
}

// ---------------------------------------------------------------------------
// out = A @ Wo^T + bo ; A bf16 [B*S, E], out fp32 row-major. 2x2 tile/wave.
// ---------------------------------------------------------------------------
__global__ void out_proj_kernel(const __bf16* __restrict__ A,
                                const __bf16* __restrict__ Wb,
                                const float* __restrict__ bias,
                                float* __restrict__ out) {
  const int wave = threadIdx.x >> 5;
  const int lane = threadIdx.x & 31;
  const int h = lane >> 4, l15 = lane & 15;
  const int gid = blockIdx.x * 4 + wave;
  const int tn = gid & 15;
  const int tm = gid >> 4;

  const __bf16* arow0 = A  + (size_t)(tm * 32 + l15) * E_;
  const __bf16* arow1 = arow0 + (size_t)16 * E_;
  const __bf16* brow0 = Wb + (size_t)(tn * 32 + l15) * E_;
  const __bf16* brow1 = brow0 + (size_t)16 * E_;

  v8f acc00 = vzero(), acc01 = vzero(), acc10 = vzero(), acc11 = vzero();
#pragma unroll 2
  for (int kc = 0; kc < E_ / 32; ++kc) {
    const int kb = kc * 32;
    v16bf a0 = afrag_bf(arow0 + kb, h);
    v16bf a1 = afrag_bf(arow1 + kb, h);
    v16bf b0 = bfrag_bf(brow0 + kb, h);
    v16bf b1 = bfrag_bf(brow1 + kb, h);
    acc00 = wmma_bf(a0, b0, acc00);
    acc01 = wmma_bf(a0, b1, acc01);
    acc10 = wmma_bf(a1, b0, acc10);
    acc11 = wmma_bf(a1, b1, acc11);
  }
#pragma unroll
  for (int j = 0; j < 2; ++j) {
    const int gcol = tn * 32 + j * 16 + l15;
    const float bv = bias[gcol];
#pragma unroll
    for (int i = 0; i < 2; ++i) {
      const v8f& acc = (i == 0) ? (j == 0 ? acc00 : acc01)
                                : (j == 0 ? acc10 : acc11);
#pragma unroll
      for (int r = 0; r < 8; ++r) {
        const int grow = tm * 32 + i * 16 + r + 8 * h;
        out[(size_t)grow * E_ + gcol] = acc[r] + bv;
      }
    }
  }
}

// ---------------------------------------------------------------------------
// Fused attention. Block = 4 waves = heads {hg*4 .. hg*4+3} of one
// (batch, 16-query tile); each wave owns a 16x2048 bf16 score panel in LDS
// (64KB/wave, 256KB/block of the 320KB WGP pool). Phases:
//   1) scores via WMMA, fused (s*scale+bias)*mod epilogue -> panel
//   2) row softmax (v8bf DS chunks, lane-pair per row, wave32 shfl),
//      then in-place normalization (panel := p * rinv)
//   [barrier]
//   2b) block-cooperative head-mean: the 4 resident panels are summed from
//       LDS and flushed with ONE atomic per element (2 atomics/element total
//       across the two head-groups, instead of 8)
//   3) attended = P@V via WMMA; A-fragment of P = two ds_load_b128
//   4) attended written as bf16 [B,S,E]
// ---------------------------------------------------------------------------
#define ATT_LDS_BYTES (4 * 16 * S_ * 2 + 4 * 16 * 4)

__global__ void emo_attention_kernel(const __bf16* __restrict__ Qb,
                                     const __bf16* __restrict__ Kb,
                                     const __bf16* __restrict__ Vt,
                                     const float* __restrict__ emo_ctx,
                                     const float* __restrict__ emo_bias,
                                     __bf16* __restrict__ att,
                                     float* __restrict__ mean_out) {
  extern __shared__ char smem_raw[];
  const int wave = threadIdx.x >> 5;
  const int lane = threadIdx.x & 31;
  const int h = lane >> 4, l15 = lane & 15;

  __bf16* panel = (__bf16*)smem_raw + (size_t)wave * 16 * S_;
  float*  rinvs = (float*)(smem_raw + 4 * 16 * S_ * 2) + wave * 16;

  // block -> (bt, hg, qt); wave -> head within group
  const int bid = blockIdx.x;            // B * 2 * (S/16) = 1024
  const int qt = bid & 127;
  const int hg = (bid >> 7) & 1;
  const int bt = bid >> 8;
  const int hh = hg * 4 + wave;
  const int q0 = qt * 16;
  const size_t hb = (size_t)(bt * H_ + hh);

  // multiplicative emotional modulation (uniform per batch)
  float msum = 0.f;
#pragma unroll 8
  for (int i = 0; i < 64; ++i) msum += emo_ctx[bt * 64 + i];
  const float mod = 0.5f + 0.5f / (1.f + __expf(-msum * (1.f / 64.f)));
  const float bias = emo_bias[hh];
  const float scale = 0.125f;  // 1/sqrt(D)

  // Q fragments for this 16-row tile (K-dim = D = 64 -> two 32-chunks)
  const __bf16* qrow = Qb + (hb * S_ + (size_t)(q0 + l15)) * D_;
  const v16bf aq0 = afrag_bf(qrow, h);
  const v16bf aq1 = afrag_bf(qrow + 32, h);

  // ---- phase 1: scores = (Q K^T * scale + bias) * mod  -> LDS panel (bf16)
  for (int kt = 0; kt < S_ / 16; ++kt) {
    const __bf16* krow = Kb + (hb * S_ + (size_t)(kt * 16 + l15)) * D_;
    v16bf b0 = bfrag_bf(krow, h);
    v16bf b1 = bfrag_bf(krow + 32, h);
    v8f acc = vzero();
    acc = wmma_bf(aq0, b0, acc);
    acc = wmma_bf(aq1, b1, acc);
#pragma unroll
    for (int r = 0; r < 8; ++r)
      panel[(r + 8 * h) * S_ + kt * 16 + l15] = f2bf((acc[r] * scale + bias) * mod);
  }

  // ---- phase 2: row softmax (lane pair per row, 1024 cols each)
  const int row = lane >> 1, hc = lane & 1;
  __bf16* prow = panel + row * S_ + hc * 1024;
  float mx = -3.0e38f;
#pragma unroll 4
  for (int c = 0; c < 128; ++c) {
    v8bf v = ld8(prow + 8 * c);
#pragma unroll
    for (int e = 0; e < 8; ++e) mx = fmaxf(mx, bf2f(v[e]));
  }
  mx = fmaxf(mx, __shfl_xor(mx, 1, 32));
  float sum = 0.f;
#pragma unroll 2
  for (int c = 0; c < 128; ++c) {
    v8bf v = ld8(prow + 8 * c), o;
#pragma unroll
    for (int e = 0; e < 8; ++e) {
      float p = __expf(bf2f(v[e]) - mx);
      o[e] = f2bf(p);
      sum += p;
    }
    st8(prow + 8 * c, o);
  }
  sum += __shfl_xor(sum, 1, 32);
  if (hc == 0) rinvs[row] = 1.f / sum;   // LDS same-wave ops are in-order

  // in-place normalization: panel := softmax probabilities (bf16)
  const float rinv = rinvs[row];
#pragma unroll 2
  for (int c = 0; c < 128; ++c) {
    v8bf v = ld8(prow + 8 * c), o;
#pragma unroll
    for (int e = 0; e < 8; ++e) o[e] = f2bf(bf2f(v[e]) * rinv);
    st8(prow + 8 * c, o);
  }

  __syncthreads();   // all 4 panels of this head-group now hold probs

  // ---- phase 2b: block-cooperative head-mean flush.
  // 128 threads cover 16 rows x 2048 cols; each sums the 4 resident panels
  // from LDS and issues one fp32 atomic (x2 head-groups = 2 atomics/elem).
  {
    const int tid = threadIdx.x;
    const int mr = tid >> 3;             // row 0..15
    const int cb = (tid & 7) * 256;      // col base, 8 threads per row
    const __bf16* base = (const __bf16*)smem_raw;
    float* mrowp = mean_out + ((size_t)bt * S_ + q0 + mr) * S_;
#pragma unroll 2
    for (int c = 0; c < 32; ++c) {
      const int col = cb + 8 * c;
      v8bf x0 = ld8(base + 0 * 16 * S_ + mr * S_ + col);
      v8bf x1 = ld8(base + 1 * 16 * S_ + mr * S_ + col);
      v8bf x2 = ld8(base + 2 * 16 * S_ + mr * S_ + col);
      v8bf x3 = ld8(base + 3 * 16 * S_ + mr * S_ + col);
#pragma unroll
      for (int e = 0; e < 8; ++e) {
        const float s4 = (bf2f(x0[e]) + bf2f(x1[e])) + (bf2f(x2[e]) + bf2f(x3[e]));
        __hip_atomic_fetch_add(&mrowp[col + e], s4 * (1.f / (float)H_),
                               __ATOMIC_RELAXED, __HIP_MEMORY_SCOPE_AGENT);
      }
    }
  }

  // ---- phase 3: attended = P @ V  (V stored transposed [B,H,D,S]).
  // A-fragment of P = two contiguous ds_load_b128 per iteration.
  v8f accv[4];
#pragma unroll
  for (int j = 0; j < 4; ++j) accv[j] = vzero();

  const __bf16* parow = panel + l15 * S_;
  for (int kc = 0; kc < S_ / 32; ++kc) {
    const int kb = kc * 32;
    v16bf ap = afrag_bf(parow + kb, h);
#pragma unroll
    for (int j = 0; j < 4; ++j) {
      const __bf16* vrow = Vt + (hb * D_ + (size_t)(j * 16 + l15)) * S_ + kb;
      v16bf bv = bfrag_bf(vrow, h);
      accv[j] = wmma_bf(ap, bv, accv[j]);
    }
  }

  // ---- phase 4: write attended directly in [B,S,E] layout (bf16)
#pragma unroll
  for (int j = 0; j < 4; ++j)
#pragma unroll
    for (int r = 0; r < 8; ++r) {
      const int s = q0 + r + 8 * h;
      att[((size_t)bt * S_ + s) * E_ + hh * 64 + j * 16 + l15] = f2bf(accv[j][r]);
    }
}

// ---------------------------------------------------------------------------
// Host launcher. Workspace layout (~50MB, regions reused once dead):
//   [0)     Wq|Wk|Wv|Wo bf16                      2MB
//   [2MB)   Xq Xk Xv bf16 inputs (8MB each)      24MB   (dead after proj)
//   [26MB)  Qb Kb Vt bf16 (8MB each)             24MB
//   Att bf16 [B,S,E] 8MB  -> overlaps Xq region (dead by then)
// d_out = [output fp32 (B,S,E)] ++ [attn_mean fp32 (B,S,S)]
// ---------------------------------------------------------------------------
extern "C" void kernel_launch(void* const* d_in, const int* in_sizes, int n_in,
                              void* d_out, int out_size, void* d_ws, size_t ws_size,
                              hipStream_t stream) {
  (void)in_sizes; (void)n_in; (void)out_size; (void)ws_size;

  const float* query    = (const float*)d_in[0];
  const float* key      = (const float*)d_in[1];
  const float* value    = (const float*)d_in[2];
  const float* emo_ctx  = (const float*)d_in[3];
  const float* Wq       = (const float*)d_in[4];
  const float* bq       = (const float*)d_in[5];
  const float* Wk       = (const float*)d_in[6];
  const float* bk       = (const float*)d_in[7];
  const float* Wv       = (const float*)d_in[8];
  const float* bv       = (const float*)d_in[9];
  const float* Wo       = (const float*)d_in[10];
  const float* bo       = (const float*)d_in[11];
  const float* emo_bias = (const float*)d_in[12];

  char* ws = (char*)d_ws;
  const size_t WSZ = (size_t)E_ * E_ * 2;               // 512KB per weight
  const size_t XSZ = (size_t)B_ * S_ * E_ * 2;          // 8MB per activation
  __bf16* WqB = (__bf16*)(ws + 0 * WSZ);
  __bf16* WkB = (__bf16*)(ws + 1 * WSZ);
  __bf16* WvB = (__bf16*)(ws + 2 * WSZ);
  __bf16* WoB = (__bf16*)(ws + 3 * WSZ);
  __bf16* Xq  = (__bf16*)(ws + 4 * WSZ);
  __bf16* Xk  = (__bf16*)(ws + 4 * WSZ + 1 * XSZ);
  __bf16* Xv  = (__bf16*)(ws + 4 * WSZ + 2 * XSZ);
  __bf16* Qb  = (__bf16*)(ws + 4 * WSZ + 3 * XSZ);
  __bf16* Kb  = (__bf16*)(ws + 4 * WSZ + 4 * XSZ);
  __bf16* Vt  = (__bf16*)(ws + 4 * WSZ + 5 * XSZ);
  __bf16* Att = Xq;  // Xq is dead once Qb exists; attention writes Att here

  float* out      = (float*)d_out;
  float* mean_out = out + (size_t)B_ * S_ * E_;

  // zero the attn-mean region (accumulated via atomics below)
  hipMemsetAsync(mean_out, 0, (size_t)B_ * S_ * S_ * sizeof(float), stream);

  // fp32 -> bf16: weights and activations
  const int nw4 = E_ * E_ / 4;            // 65536
  const int nx4 = B_ * S_ * E_ / 4;       // 1048576
  cvt_bf16_kernel<<<dim3(nw4 / 256), dim3(256), 0, stream>>>(Wq, WqB, nw4);
  cvt_bf16_kernel<<<dim3(nw4 / 256), dim3(256), 0, stream>>>(Wk, WkB, nw4);
  cvt_bf16_kernel<<<dim3(nw4 / 256), dim3(256), 0, stream>>>(Wv, WvB, nw4);
  cvt_bf16_kernel<<<dim3(nw4 / 256), dim3(256), 0, stream>>>(Wo, WoB, nw4);
  cvt_bf16_kernel<<<dim3(nx4 / 256), dim3(256), 0, stream>>>(query, Xq, nx4);
  cvt_bf16_kernel<<<dim3(nx4 / 256), dim3(256), 0, stream>>>(key,   Xk, nx4);
  cvt_bf16_kernel<<<dim3(nx4 / 256), dim3(256), 0, stream>>>(value, Xv, nx4);

  // Q/K/V projections: 4096 32x32 tiles, 4 waves (128 thr) per block
  const int proj_blocks = (B_ * S_ / 32) * (E_ / 32) / 4;   // 1024
  proj_qkv_kernel<0><<<dim3(proj_blocks), dim3(128), 0, stream>>>(Xq, WqB, bq, Qb);
  proj_qkv_kernel<0><<<dim3(proj_blocks), dim3(128), 0, stream>>>(Xk, WkB, bk, Kb);
  proj_qkv_kernel<1><<<dim3(proj_blocks), dim3(128), 0, stream>>>(Xv, WvB, bv, Vt);

  // fused attention: B*2*(S/16)=1024 blocks x 4 waves (head-groups of 4),
  // 256KB+256B dynamic LDS per block (within the 320KB WGP pool)
  hipFuncSetAttribute(reinterpret_cast<const void*>(emo_attention_kernel),
                      hipFuncAttributeMaxDynamicSharedMemorySize, ATT_LDS_BYTES);
  emo_attention_kernel<<<dim3(1024), dim3(128), ATT_LDS_BYTES, stream>>>(
      Qb, Kb, Vt, emo_ctx, emo_bias, Att, mean_out);

  // output projection -> fp32 d_out
  out_proj_kernel<<<dim3(proj_blocks), dim3(128), 0, stream>>>(Att, WoB, bo, out);
}
